// seq2seqModel_46583215292670
// MI455X (gfx1250) — matrix operand order repaired and checked by reading, not compiled, not run on hardware
//
#include <hip/hip_runtime.h>

// ---------------------------------------------------------------------------
// Mamba (B=2, L=2048) for MI455X / gfx1250, wave32.
//  - GEMMs (in_proj / x_proj / out_proj) via v_wmma_f32_16x16x32_bf16 with the
//    K-loop fully unrolled (templated K): 4 / 8 chained WMMAs, fp32 accum.
//  - Whole 16xK A tile staged in LDS once (single barrier pair), then the
//    WMMA chain runs barrier-free; fragments are vector loads (ds_load_b128 x2
//    for A, global_load_b128 x2 for B) in contiguous k order (k = 16*half + 0..15).
//  - Selective scan: 512 channels x 16 states = 256 wave32 in parallel;
//    per-step 16-lane shfl_xor reduction for y = sum_s h*C.
//  - ~8 GF total, fits in L2; latency-chain bound -> maximize parallel waves.
// ---------------------------------------------------------------------------

#define BB   2
#define LL   2048
#define CINC 20
#define DM   128
#define DI   256
#define DS   16
#define DTRN 8
#define DC   4
#define NL   6
#define NC   6
#define ROWS (BB*LL)     // 4096 token rows
#define XPN  48          // x_proj output padded 40 -> 48 (3 n-tiles)
#define EPSV 1e-5f

typedef __attribute__((ext_vector_type(16))) __bf16 v16bf;
typedef __attribute__((ext_vector_type(8)))  __bf16 v8bf;
typedef __attribute__((ext_vector_type(8)))  float  v8f;

__device__ __forceinline__ __bf16 f2bf(float f) {
  unsigned u = __float_as_uint(f);
  u += 0x7FFFu + ((u >> 16) & 1u);          // round-to-nearest-even
  unsigned short h = (unsigned short)(u >> 16);
  return __builtin_bit_cast(__bf16, h);
}
__device__ __forceinline__ float sigmoidf_(float x) { return 1.f / (1.f + __expf(-x)); }
__device__ __forceinline__ float siluf_(float x)    { return x * sigmoidf_(x); }
__device__ __forceinline__ float softplusf_(float x){ return (x > 20.f) ? x : log1pf(__expf(x)); }

// --------------------------- embed: rmsnorm + in2m + silu -------------------
__global__ __launch_bounds__(DM)
void k_embed(const float* __restrict__ xin, const float* __restrict__ in_norm_w,
             const float* __restrict__ w, const float* __restrict__ b,
             float* __restrict__ hs) {
  const int row = blockIdx.x, dm = threadIdx.x;
  const float* xr = xin + (size_t)row * CINC;
  float ss = 0.f;
  #pragma unroll
  for (int c = 0; c < CINC; ++c) { float v = xr[c]; ss += v * v; }
  const float rms = rsqrtf(ss / CINC + EPSV);
  float acc = b[dm];
  #pragma unroll
  for (int c = 0; c < CINC; ++c) acc += xr[c] * rms * in_norm_w[c] * w[dm * CINC + c];
  hs[(size_t)row * DM + dm] = siluf_(acc);
}

// ---------------- residual update + rmsnorm + bf16 cast ---------------------
__global__ __launch_bounds__(DM)
void k_pre(const float* __restrict__ hs, float* __restrict__ res,
           const float* __restrict__ norm_w, __bf16* __restrict__ xn, int first) {
  __shared__ float red[DM];
  const int row = blockIdx.x, t = threadIdx.x;
  const size_t idx = (size_t)row * DM + t;
  float v = hs[idx] + (first ? 0.f : res[idx]);
  res[idx] = v;
  red[t] = v * v; __syncthreads();
  for (int off = DM / 2; off > 0; off >>= 1) {
    if (t < off) red[t] += red[t + off];
    __syncthreads();
  }
  const float rms = rsqrtf(red[0] / DM + EPSV);
  xn[idx] = f2bf(v * rms * norm_w[t]);
}

// ------------------- fp32 weight (N x K) -> bf16 padded (Npad x K) ----------
// 2D grid (row = blockIdx.y) -> no runtime integer division.
__global__ __launch_bounds__(128)
void k_w2bf(const float* __restrict__ src, __bf16* __restrict__ dst,
            int N, int K) {
  const int r = blockIdx.y;
  const int c = blockIdx.x * blockDim.x + threadIdx.x;
  if (c >= K) return;
  dst[(size_t)r * K + c] = (r < N) ? f2bf(src[(size_t)r * K + c]) : f2bf(0.f);
}

// --------------------------- WMMA GEMM: C = A @ W^T -------------------------
// A: (M x K) bf16 row-major; W: (N x K) bf16 row-major (so B[k][n] = W[n][k]).
// One 16x16 C tile per wave; block stages its whole 16xK A tile in LDS once,
// then the templated-K loop fully unrolls into a barrier-free WMMA chain.
template <int K>
__global__ __launch_bounds__(256)
void k_gemm_bf16(const __bf16* __restrict__ A, const __bf16* __restrict__ W,
                 float* __restrict__ C, int N) {
  constexpr int LSTR = K + 8;   // bf16 elems; rows stay 16B-aligned, banks spread
  __shared__ __bf16 As[16 * LSTR];
  const int lane = threadIdx.x & 31;
  const int wave = threadIdx.x >> 5;
  const int wpb  = blockDim.x >> 5;
  const int m0   = blockIdx.y * 16;
  const int n0   = (blockIdx.x * wpb + wave) * 16;
  const int half = lane >> 4;                    // K-half select: k = 16*half + 0..15 (contiguous order)
  const int l16  = lane & 15;

  // Stage the full 16xK A tile cooperatively, 4B (element-pair) granules.
  constexpr int KH = K / 2;                      // uints per tile row
  for (int idx = threadIdx.x; idx < 16 * KH; idx += blockDim.x) {
    const int r = idx / KH, c = idx & (KH - 1);
    reinterpret_cast<unsigned int*>(As)[r * (LSTR / 2) + c] =
        *reinterpret_cast<const unsigned int*>(A + (size_t)(m0 + r) * K + 2 * c);
  }
  __syncthreads();

  v8f acc = {};
  #pragma unroll
  for (int k0 = 0; k0 < K; k0 += 32) {
    // A fragment (16-bit 16x32): per lane two contiguous 16B runs:
    //   elems 0..7  <- K = k0 + 16*half + 0..7
    //   elems 8..15 <- K = k0 + 16*half + 8..15
    const __bf16* arow = As + l16 * LSTR + k0 + 16 * half;
    const v8bf alo = *reinterpret_cast<const v8bf*>(arow);
    const v8bf ahi = *reinterpret_cast<const v8bf*>(arow + 8);
    const v16bf af = __builtin_shufflevector(alo, ahi,
        0, 1, 2, 3, 4, 5, 6, 7, 8, 9, 10, 11, 12, 13, 14, 15);

    // B fragment (16-bit 32x16): lane n = l16, K = k0 + 16*half + 0..15 contiguous.
    const __bf16* wrow = W + (size_t)(n0 + l16) * K + k0 + 16 * half;
    if (k0 + 32 < K) __builtin_prefetch((const void*)(wrow + 32), 0, 2);
    const v16bf wf = *reinterpret_cast<const v16bf*>(wrow);

    acc = __builtin_amdgcn_wmma_f32_16x16x32_bf16(false, af, false, wf,
                                                  (short)0, acc, false, false);
  }
  // C/D layout: VGPR r -> M = r + 8*half, N = lane&15.
  #pragma unroll
  for (int r = 0; r < 8; ++r)
    C[(size_t)(m0 + r + 8 * half) * N + n0 + l16] = acc[r];
}

// ------------------- depthwise causal conv (DC=4) + silu --------------------
__global__ __launch_bounds__(DI)
void k_conv(const float* __restrict__ xz, const float* __restrict__ cw,
            const float* __restrict__ cb, float* __restrict__ u32,
            __bf16* __restrict__ u16) {
  const int row = blockIdx.x, d = threadIdx.x;
  const int b = row / LL, l = row - b * LL;      // LL is a power of two
  float acc = cb[d];
  #pragma unroll
  for (int j = 0; j < DC; ++j) {
    const int ll = l - (DC - 1) + j;
    if (ll >= 0) acc += xz[((size_t)(b * LL + ll)) * (2 * DI) + d] * cw[d * DC + j];
  }
  const float uu = siluf_(acc);
  u32[(size_t)row * DI + d] = uu;
  u16[(size_t)row * DI + d] = f2bf(uu);
}

// --------------------------- dt = softplus(dtr @ dtw^T + b) -----------------
__global__ __launch_bounds__(DI)
void k_dt(const float* __restrict__ xdbl, const float* __restrict__ dtw,
          const float* __restrict__ dtb, float* __restrict__ dt) {
  const int row = blockIdx.x, d = threadIdx.x;
  float acc = dtb[d];
  #pragma unroll
  for (int r = 0; r < DTRN; ++r) acc += xdbl[(size_t)row * XPN + r] * dtw[d * DTRN + r];
  dt[(size_t)row * DI + d] = softplusf_(acc);
}

// ------------------- selective scan + D skip + silu(z) gating ---------------
// 16 lanes per channel (one per state s); 16 channels per 256-thread block.
__global__ __launch_bounds__(256)
void k_scan(const float* __restrict__ dt, const float* __restrict__ u32,
            const float* __restrict__ xdbl, const float* __restrict__ xz,
            const float* __restrict__ A_log, const float* __restrict__ Dp,
            __bf16* __restrict__ gated) {
  const int grp = threadIdx.x >> 4;
  const int s   = threadIdx.x & 15;
  const int ch  = blockIdx.x * 16 + grp;        // 0..511
  const int b   = ch >> 8;                      // / DI
  const int d   = ch & (DI - 1);
  const float a  = -__expf(A_log[d * DS + s]);
  const float Dv = Dp[d];
  float h = 0.f;
  for (int t = 0; t < LL; ++t) {
    const size_t row = (size_t)b * LL + t;
    const float dtv = dt[row * DI + d];
    const float uu  = u32[row * DI + d];
    const float Bs  = xdbl[row * XPN + DTRN + s];
    const float Cs  = xdbl[row * XPN + DTRN + DS + s];
    h = __expf(dtv * a) * h + (dtv * uu) * Bs;
    float p = h * Cs;
    p += __shfl_xor(p, 8, 16);
    p += __shfl_xor(p, 4, 16);
    p += __shfl_xor(p, 2, 16);
    p += __shfl_xor(p, 1, 16);
    if (s == 0) {
      const float y = p + uu * Dv;
      const float z = xz[row * (2 * DI) + DI + d];
      gated[row * DI + d] = f2bf(y * siluf_(z));
    }
  }
}

// ------------------- final rmsnorm + head (last token) + softmax ------------
__global__ __launch_bounds__(DM)
void k_head(const float* __restrict__ hs, const float* __restrict__ res,
            const float* __restrict__ nfw, const float* __restrict__ hw,
            const float* __restrict__ hb, float* __restrict__ out) {
  __shared__ float red[DM];
  __shared__ float hv[DM];
  __shared__ float lo[NC];
  const int b = blockIdx.x, t = threadIdx.x;
  const size_t idx = ((size_t)b * LL + (LL - 1)) * DM + t;
  float v = hs[idx] + res[idx];
  red[t] = v * v; __syncthreads();
  for (int off = DM / 2; off > 0; off >>= 1) {
    if (t < off) red[t] += red[t + off];
    __syncthreads();
  }
  hv[t] = v * rsqrtf(red[0] / DM + EPSV) * nfw[t];
  __syncthreads();
  if (t < NC) {
    float acc = hb[t];
    for (int k = 0; k < DM; ++k) acc += hv[k] * hw[t * DM + k];
    lo[t] = acc;
  }
  __syncthreads();
  if (t == 0) {
    float mx = lo[0];
    for (int c = 1; c < NC; ++c) mx = fmaxf(mx, lo[c]);
    float e[NC], sum = 0.f;
    for (int c = 0; c < NC; ++c) { e[c] = __expf(lo[c] - mx); sum += e[c]; }
    for (int c = 0; c < NC; ++c) out[b * NC + c] = e[c] / sum;
  }
}

// ---------------------------------------------------------------------------
extern "C" void kernel_launch(void* const* d_in, const int* in_sizes, int n_in,
                              void* d_out, int out_size, void* d_ws, size_t ws_size,
                              hipStream_t stream) {
  (void)in_sizes; (void)n_in; (void)out_size; (void)ws_size;
  const float* input_ids  = (const float*)d_in[0];
  const float* in_norm_w  = (const float*)d_in[1];
  const float* in2m_w     = (const float*)d_in[2];
  const float* in2m_b     = (const float*)d_in[3];
  const float* norm_w     = (const float*)d_in[4];
  const float* in_proj_w  = (const float*)d_in[5];
  const float* conv_w     = (const float*)d_in[6];
  const float* conv_b     = (const float*)d_in[7];
  const float* x_proj_w   = (const float*)d_in[8];
  const float* dt_proj_w  = (const float*)d_in[9];
  const float* dt_proj_b  = (const float*)d_in[10];
  const float* A_log      = (const float*)d_in[11];
  const float* D_param    = (const float*)d_in[12];
  const float* out_proj_w = (const float*)d_in[13];
  const float* norm_f_w   = (const float*)d_in[14];
  const float* head_w     = (const float*)d_in[15];
  const float* head_b     = (const float*)d_in[16];

  char* p = (char*)d_ws;
  auto take = [&](size_t bytes) -> void* {
    void* r = (void*)p;
    p += (bytes + 255) & ~(size_t)255;
    return r;
  };
  float*  hs     = (float*) take((size_t)ROWS * DM * 4);
  float*  res    = (float*) take((size_t)ROWS * DM * 4);
  __bf16* xn16   = (__bf16*)take((size_t)ROWS * DM * 2);
  __bf16* win16  = (__bf16*)take((size_t)2 * DI * DM * 2);
  __bf16* wxp16  = (__bf16*)take((size_t)XPN * DI * 2);
  __bf16* wout16 = (__bf16*)take((size_t)DM * DI * 2);
  float*  xz     = (float*) take((size_t)ROWS * 2 * DI * 4);
  float*  u32    = (float*) take((size_t)ROWS * DI * 4);
  __bf16* u16    = (__bf16*)take((size_t)ROWS * DI * 2);
  float*  xdbl   = (float*) take((size_t)ROWS * XPN * 4);
  float*  dtb    = (float*) take((size_t)ROWS * DI * 4);
  __bf16* g16    = (__bf16*)take((size_t)ROWS * DI * 2);

  k_embed<<<ROWS, DM, 0, stream>>>(input_ids, in_norm_w, in2m_w, in2m_b, hs);

  for (int i = 0; i < NL; ++i) {
    k_w2bf<<<dim3(DM / 128, 2 * DI), 128, 0, stream>>>(
        in_proj_w + (size_t)i * 2 * DI * DM, win16, 2 * DI, DM);
    k_w2bf<<<dim3(DI / 128, XPN), 128, 0, stream>>>(
        x_proj_w + (size_t)i * (DTRN + 2 * DS) * DI, wxp16, DTRN + 2 * DS, DI);
    k_w2bf<<<dim3(DI / 128, DM), 128, 0, stream>>>(
        out_proj_w + (size_t)i * DM * DI, wout16, DM, DI);

    k_pre<<<ROWS, DM, 0, stream>>>(hs, res, norm_w + (size_t)i * DM, xn16, i == 0);

    // in_proj: (4096x128)@(128x512); 32 n-tiles, 8 waves/block -> grid.x=4
    k_gemm_bf16<DM><<<dim3(4, ROWS / 16), 256, 0, stream>>>(xn16, win16, xz, 2 * DI);

    k_conv<<<ROWS, DI, 0, stream>>>(xz, conv_w + (size_t)i * DI * DC, conv_b + (size_t)i * DI, u32, u16);

    // x_proj: (4096x256)@(256x48pad); 3 n-tiles, 3 waves/block
    k_gemm_bf16<DI><<<dim3(1, ROWS / 16), 96, 0, stream>>>(u16, wxp16, xdbl, XPN);

    k_dt<<<ROWS, DI, 0, stream>>>(xdbl, dt_proj_w + (size_t)i * DI * DTRN, dt_proj_b + (size_t)i * DI, dtb);

    k_scan<<<(BB * DI) / 16, 256, 0, stream>>>(dtb, u32, xdbl, xz,
                                               A_log + (size_t)i * DI * DS,
                                               D_param + (size_t)i * DI, g16);

    // out_proj: (4096x256)@(256x128); 8 n-tiles, 8 waves/block
    k_gemm_bf16<DI><<<dim3(1, ROWS / 16), 256, 0, stream>>>(g16, wout16, hs, DM);
  }

  k_head<<<BB, DM, 0, stream>>>(hs, res, norm_f_w, head_w, head_b, (float*)d_out);
}
